// Head_36644660969572
// MI455X (gfx1250) — matrix operand order repaired and checked by reading, not compiled, not run on hardware
//
#include <hip/hip_runtime.h>

typedef __bf16 bf16;
typedef __attribute__((ext_vector_type(16))) __bf16 v16bf;
typedef __attribute__((ext_vector_type(8)))  __bf16 v8bf;
typedef __attribute__((ext_vector_type(2)))  __bf16 v2bf;
typedef __attribute__((ext_vector_type(8)))  float   v8f;
typedef __attribute__((ext_vector_type(4)))  int     v4i;

#define B_ 8
#define T_ 2048
#define C_ 2048
#define H_ 128

#if defined(__has_builtin)
#  if __has_builtin(__builtin_amdgcn_global_load_async_to_lds_b128)
#    define HAVE_ASYNC_LDS 1
#  endif
#endif
#ifndef HAVE_ASYNC_LDS
#  define HAVE_ASYNC_LDS 0
#endif

#if HAVE_ASYNC_LDS
#  if __has_builtin(__builtin_amdgcn_s_wait_asynccnt)
#    define ASYNC_WAIT() __builtin_amdgcn_s_wait_asynccnt(0)
#  else
#    define ASYNC_WAIT() asm volatile("s_wait_asynccnt 0x0" ::: "memory")
#  endif
#endif

// ---------- helpers ----------

__device__ __forceinline__ bf16 f2bf(float f) { return (bf16)f; }

__device__ __forceinline__ v2bf f2bf2(float a, float b) {
#if defined(__has_builtin) && __has_builtin(__builtin_amdgcn_cvt_pk_bf16_f32)
  return __builtin_amdgcn_cvt_pk_bf16_f32(a, b);
#else
  v2bf r; r[0] = (bf16)a; r[1] = (bf16)b; return r;
#endif
}

__device__ __forceinline__ v8f wmma_bf16(v16bf a, v16bf b, v8f c) {
  // D = A(16x32 bf16) * B(32x16 bf16) + C(16x16 f32)
  return __builtin_amdgcn_wmma_f32_16x16x32_bf16(
      /*neg_a=*/false, a, /*neg_b=*/false, b,
      /*c_mod=*/(short)0, c, /*reuse_a=*/false, /*reuse_b=*/false);
}

// A-matrix fragment (16x32, 16-bit). Per ISA layout: lanes 0-15 hold
// K = {0..7, 16..23}; lanes 16-31 hold K = {8..15, 24..31}. rowBase points at
// this lane's row (row = lane & 15), contraction-contiguous.
__device__ __forceinline__ v16bf load_a_frag(const bf16* rowBase, int lh) {
  const v8bf* p0 = (const v8bf*)(rowBase + lh * 8);
  const v8bf* p1 = (const v8bf*)(rowBase + 16 + lh * 8);
  v8bf lo = *p0, hi = *p1;
  v16bf r;
#pragma unroll
  for (int i = 0; i < 8; ++i) { r[i] = lo[i]; r[i + 8] = hi[i]; }
  return r;
}

// B-matrix fragment (32x16, 16-bit), source supplied "column-major" so that
// column n of B (over the contraction dim) is a contiguous row.
// lanes 0-15 hold K={0..7, 16..23}, lanes 16-31 hold K={8..15, 24..31}.
__device__ __forceinline__ v16bf load_b_frag(const bf16* rowBase, int lh) {
  const v8bf* p0 = (const v8bf*)(rowBase + lh * 8);
  const v8bf* p1 = (const v8bf*)(rowBase + 16 + lh * 8);
  v8bf lo = *p0, hi = *p1;
  v16bf r;
#pragma unroll
  for (int i = 0; i < 8; ++i) { r[i] = lo[i]; r[i + 8] = hi[i]; }
  return r;
}

// ---------- stage 0: W [C,H] f32 -> Wt [H,C] bf16 ----------

__global__ void wt_convert_kernel(const float* __restrict__ Wq,
                                  const float* __restrict__ Wk,
                                  const float* __restrict__ Wv,
                                  bf16* __restrict__ Wtq,
                                  bf16* __restrict__ Wtk,
                                  bf16* __restrict__ Wtv) {
  int idx = blockIdx.x * blockDim.x + threadIdx.x;
  if (idx >= C_ * H_) return;
  int c = idx / H_, h = idx % H_;
  const float* W = (blockIdx.y == 0) ? Wq : (blockIdx.y == 1) ? Wk : Wv;
  bf16* Wt = (blockIdx.y == 0) ? Wtq : (blockIdx.y == 1) ? Wtk : Wtv;
  Wt[(size_t)h * C_ + c] = f2bf(W[idx]);
}

// ---------- stage 1: fused q/k/v projection via WMMA ----------
// One wave per 16-row M tile; computes the full N=128 for q, k, v reusing the
// x A-fragment (x is read exactly once). q pre-scaled by 1/sqrt(C).
// q,k stored [B*T][H]; v stored TRANSPOSED [B][H][T] so attention can read
// V^T B-fragments contiguously from global (and each lane's 8 consecutive
// t-values pack into one b128 store).

__global__ void __launch_bounds__(128) proj_qkv_kernel(
    const float* __restrict__ x,
    const bf16* __restrict__ Wtq, const bf16* __restrict__ Wtk,
    const bf16* __restrict__ Wtv,
    bf16* __restrict__ qb, bf16* __restrict__ kb, bf16* __restrict__ vtb) {
  const int lane = threadIdx.x & 31;
  const int wid  = threadIdx.x >> 5;
  const int m    = lane & 15;
  const int lh   = lane >> 4;
  const int row0 = (blockIdx.x * 4 + wid) * 16;
  const float* xrow = x + (size_t)(row0 + m) * C_;

  v8f vzero = {};
  v8f accQ[8], accK[8], accV[8];
#pragma unroll
  for (int i = 0; i < 8; ++i) { accQ[i] = vzero; accK[i] = vzero; accV[i] = vzero; }

  for (int kc = 0; kc < C_ / 32; ++kc) {
    // A fragment: x row slice, f32 -> bf16 via packed convert (A layout)
    const float* ap0 = xrow + kc * 32 + lh * 8;
    const float* ap1 = ap0 + 16;
    v16bf a;
#pragma unroll
    for (int i = 0; i < 4; ++i) {
      v2bf lo = f2bf2(ap0[2 * i], ap0[2 * i + 1]);
      v2bf hi = f2bf2(ap1[2 * i], ap1[2 * i + 1]);
      a[2 * i]     = lo[0]; a[2 * i + 1]     = lo[1];
      a[8 + 2 * i] = hi[0]; a[8 + 2 * i + 1] = hi[1];
    }
#pragma unroll
    for (int nt = 0; nt < 8; ++nt) {
      const size_t wrow = (size_t)(nt * 16 + m) * C_ + (size_t)kc * 32;
      accQ[nt] = wmma_bf16(a, load_b_frag(Wtq + wrow, lh), accQ[nt]);
      accK[nt] = wmma_bf16(a, load_b_frag(Wtk + wrow, lh), accK[nt]);
      accV[nt] = wmma_bf16(a, load_b_frag(Wtv + wrow, lh), accV[nt]);
    }
  }

  const float qscale = 0.02209708691207961f;  // C^-0.5, C = 2048
  const int brow = row0 / T_;                 // batch (T_ % 16 == 0)
  const int t0   = row0 % T_;
#pragma unroll
  for (int nt = 0; nt < 8; ++nt) {
#pragma unroll
    for (int v = 0; v < 8; ++v) {
      // C/D layout: row = v + 8*half, col = nt*16 + (lane&15)
      size_t off = (size_t)(row0 + v + 8 * lh) * H_ + nt * 16 + m;
      qb[off] = f2bf(accQ[nt][v] * qscale);
      kb[off] = f2bf(accK[nt][v]);
    }
    // V transposed: lane holds 8 consecutive t for column h -> one b128 store
    v8bf pv;
#pragma unroll
    for (int v = 0; v < 8; ++v) pv[v] = f2bf(accV[nt][v]);
    *(v8bf*)(vtb + ((size_t)brow * H_ + nt * 16 + m) * T_ + t0 + 8 * lh) = pv;
  }
}

// ---------- stage 2: causal flash attention ----------
// Block = 4 waves; each wave owns 16 query rows. K tile (32x128) staged in
// LDS via async global->LDS loads (shared by all 4 waves); V^T read straight
// from global (pre-transposed); P transposed through a small LDS scratch.

__global__ void __launch_bounds__(128) attn_kernel(
    const bf16* __restrict__ qb, const bf16* __restrict__ kb,
    const bf16* __restrict__ vtb, float* __restrict__ out) {
  __shared__ bf16 Kt[32][H_];        // 8 KB : K tile, row-major
  __shared__ bf16 Pt[4][16][32];     // 4 KB : per-wave P transpose scratch

  const int b    = blockIdx.y;
  const int qBlk = blockIdx.x;       // 64 queries per block
  const int wid  = threadIdx.x >> 5;
  const int lane = threadIdx.x & 31;
  const int m    = lane & 15;
  const int lh   = lane >> 4;
  const int qBase = qBlk * 64 + wid * 16;

  // Q fragments (scale already folded in), kept resident
  const bf16* qrow = qb + ((size_t)b * T_ + qBase + m) * H_;
  v16bf qf[4];
#pragma unroll
  for (int kc = 0; kc < 4; ++kc) qf[kc] = load_a_frag(qrow + kc * 32, lh);

  v8f vzero = {};
  v8f o[8];
#pragma unroll
  for (int i = 0; i < 8; ++i) o[i] = vzero;
  float rowMax[8], rowSum[8];
#pragma unroll
  for (int v = 0; v < 8; ++v) { rowMax[v] = -__builtin_inff(); rowSum[v] = 0.f; }

  const bf16* vbase = vtb + (size_t)b * H_ * T_;

  const int nTiles = qBlk * 2 + 2;   // uniform across waves -> barriers legal
  for (int t = 0; t < nTiles; ++t) {
    const int kBase = t * 32;
    __syncthreads();                 // previous iteration done reading Kt/Pt
    {
      const int srow = threadIdx.x >> 2;        // 0..31 key row
      const int cb   = (threadIdx.x & 3) * 32;  // head-dim chunk
      const bf16* ks = kb + ((size_t)b * T_ + kBase + srow) * H_ + cb;
      bf16* kd = &Kt[srow][cb];
#if HAVE_ASYNC_LDS
#pragma unroll
      for (int i = 0; i < 4; ++i)
        __builtin_amdgcn_global_load_async_to_lds_b128(
            (v4i*)(ks + i * 8), (v4i*)(kd + i * 8), 0, 0);
      ASYNC_WAIT();
#else
#pragma unroll
      for (int i = 0; i < 4; ++i)
        *(v8bf*)(kd + i * 8) = *(const v8bf*)(ks + i * 8);
#endif
    }
    __syncthreads();

    // S = Q * K^T : two 16x16 f32 tiles (keys n=0..15 and 16..31)
    v8f s0 = vzero, s1 = vzero;
#pragma unroll
    for (int kc = 0; kc < 4; ++kc) {
      s0 = wmma_bf16(qf[kc], load_b_frag(&Kt[m][kc * 32], lh), s0);
      s1 = wmma_bf16(qf[kc], load_b_frag(&Kt[16 + m][kc * 32], lh), s1);
    }

    // causal mask + online softmax (row = v + 8*lh lives across 16 lanes)
    float alpha[8], psum[8];
#pragma unroll
    for (int v = 0; v < 8; ++v) {
      const int r = qBase + v + 8 * lh;
      if (kBase + m > r)      s0[v] = -__builtin_inff();
      if (kBase + 16 + m > r) s1[v] = -__builtin_inff();
      float tm = fmaxf(s0[v], s1[v]);
#pragma unroll
      for (int off = 1; off < 16; off <<= 1)
        tm = fmaxf(tm, __shfl_xor(tm, off, 16));
      const float nm = fmaxf(rowMax[v], tm);
      alpha[v] = __expf(rowMax[v] - nm);
      rowMax[v] = nm;
      const float p0 = __expf(s0[v] - nm);
      const float p1 = __expf(s1[v] - nm);
      s0[v] = p0; s1[v] = p1;
      float ps = p0 + p1;
#pragma unroll
      for (int off = 1; off < 16; off <<= 1)
        ps += __shfl_xor(ps, off, 16);
      psum[v] = ps;
    }
#pragma unroll
    for (int v = 0; v < 8; ++v) {
      rowSum[v] = rowSum[v] * alpha[v] + psum[v];
#pragma unroll
      for (int ht = 0; ht < 8; ++ht) o[ht][v] *= alpha[v];
    }

    // P (C-layout) -> LDS -> A-fragment layout
#pragma unroll
    for (int v = 0; v < 8; ++v) {
      Pt[wid][v + 8 * lh][m]      = f2bf(s0[v]);
      Pt[wid][v + 8 * lh][m + 16] = f2bf(s1[v]);
    }
    __syncthreads();
    v16bf pf = load_a_frag(&Pt[wid][m][0], lh);

    // O += P(16x32) * V(32x128); V^T rows contiguous in global
#pragma unroll
    for (int ht = 0; ht < 8; ++ht) {
      const bf16* vrow = vbase + (size_t)(ht * 16 + m) * T_ + kBase;
      o[ht] = wmma_bf16(pf, load_b_frag(vrow, lh), o[ht]);
    }
  }

  // normalize and write f32 output
#pragma unroll
  for (int v = 0; v < 8; ++v) {
    const float inv = 1.f / rowSum[v];
    const int r = qBase + v + 8 * lh;
#pragma unroll
    for (int ht = 0; ht < 8; ++ht)
      out[((size_t)b * T_ + r) * H_ + ht * 16 + m] = o[ht][v] * inv;
  }
}

// ---------- launch ----------

extern "C" void kernel_launch(void* const* d_in, const int* in_sizes, int n_in,
                              void* d_out, int out_size, void* d_ws, size_t ws_size,
                              hipStream_t stream) {
  (void)in_sizes; (void)n_in; (void)out_size; (void)ws_size;
  const float* x  = (const float*)d_in[0];
  const float* Wq = (const float*)d_in[1];
  const float* Wk = (const float*)d_in[2];
  const float* Wv = (const float*)d_in[3];
  float* out = (float*)d_out;

  char* ws = (char*)d_ws;
  const size_t wtBytes  = (size_t)H_ * C_ * sizeof(bf16);       // 512 KB each
  const size_t qkvBytes = (size_t)B_ * T_ * H_ * sizeof(bf16);  // 4 MB each
  bf16* Wtq = (bf16*)(ws);
  bf16* Wtk = (bf16*)(ws + wtBytes);
  bf16* Wtv = (bf16*)(ws + 2 * wtBytes);
  bf16* qb  = (bf16*)(ws + 3 * wtBytes);
  bf16* kb  = (bf16*)((char*)qb + qkvBytes);
  bf16* vtb = (bf16*)((char*)kb + qkvBytes);   // [B][H][T] transposed

  // stage 0: weight transpose + bf16 convert (3 matrices via blockIdx.y)
  wt_convert_kernel<<<dim3((C_ * H_ + 255) / 256, 3), 256, 0, stream>>>(
      Wq, Wk, Wv, Wtq, Wtk, Wtv);

  // stage 1: fused q/k/v projection, 1024 waves (one 16-row tile each)
  proj_qkv_kernel<<<(B_ * T_ / 16) / 4, 128, 0, stream>>>(
      x, Wtq, Wtk, Wtv, qb, kb, vtb);

  // stage 2: flash attention, grid = (T/64, B), 4 waves per block
  attn_kernel<<<dim3(T_ / 64, B_), 128, 0, stream>>>(qb, kb, vtb, out);
}
